// ProtoViTBase_41669772705822
// MI455X (gfx1250) — matrix operand order repaired and checked by reading, not compiled, not run on hardware
//
#include <hip/hip_runtime.h>
#include <stdint.h>

#define USE_ASYNC_LDS 1

#define B_     16
#define D_     384
#define DEPTH_ 6
#define H_     6
#define HD_    64
#define P_     2000
#define NP_    4
#define N_     196
#define T_     197
#define MT_    (B_ * T_)          // 3152
#define PNPAD_ 8064               // P*NP padded to 126*64

typedef __attribute__((ext_vector_type(16))) _Float16 v16h;
typedef __attribute__((ext_vector_type(8)))  float    v8f;

__device__ __forceinline__ v8f wmma_f32_f16(v16h a, v16h b, v8f c) {
  return __builtin_amdgcn_wmma_f32_16x16x32_f16(false, a, false, b, (short)0, c,
                                                false, false);
}

__device__ __forceinline__ void wait_async0() {
#if defined(__has_builtin)
#if __has_builtin(__builtin_amdgcn_s_wait_asynccnt)
  __builtin_amdgcn_s_wait_asynccnt(0);
#else
  asm volatile("s_wait_asynccnt 0" ::: "memory");
#endif
#else
  asm volatile("s_wait_asynccnt 0" ::: "memory");
#endif
}

#if USE_ASYNC_LDS
// CDNA5 async DMA: copy 16 bytes/lane global -> LDS, tracked by ASYNCcnt.
__device__ __forceinline__ void async_b128(void* lds, const void* gptr) {
  uint32_t l = (uint32_t)(uintptr_t)lds;         // LDS aperture: addr[31:0] = offset
  uint64_t g = (uint64_t)(uintptr_t)gptr;
  asm volatile("global_load_async_to_lds_b128 %0, %1, off"
               :: "v"(l), "v"(g) : "memory");
}
#endif

// A fragment: 16x32 f16 tile, row-major, ld halves (even).
// Lanes 0-15 row m hold K 0..15 contiguous; lanes 16-31 hold K 16..31.
__device__ __forceinline__ v16h load_fragA(const _Float16* tile, int ld) {
  int lane = threadIdx.x & 31;
  const _Float16* p = tile + (lane & 15) * ld + ((lane >> 4) << 4);
  const uint32_t* q0 = (const uint32_t*)p;
  const uint32_t* q1 = (const uint32_t*)(p + 8);
  union { uint32_t u[8]; v16h h; } r;
  r.u[0] = q0[0]; r.u[1] = q0[1]; r.u[2] = q0[2]; r.u[3] = q0[3];
  r.u[4] = q1[0]; r.u[5] = q1[1]; r.u[6] = q1[2]; r.u[7] = q1[3];
  return r.h;
}

// B fragment from Bt layout [n][k]: lanes 0-15 K=0..15 contiguous, 16-31 K=16..31.
__device__ __forceinline__ v16h load_fragB(const _Float16* tile, int ld) {
  int lane = threadIdx.x & 31;
  const _Float16* p = tile + (lane & 15) * ld + ((lane >> 4) << 4);
  const uint32_t* q = (const uint32_t*)p;
  union { uint32_t u[8]; v16h h; } r;
#pragma unroll
  for (int i = 0; i < 8; ++i) r.u[i] = q[i];
  return r.h;
}

// ---------------- weight prep ----------------

__global__ __launch_bounds__(256) void k_convert(const float* __restrict__ s,
                                                 _Float16* __restrict__ d, int n) {
  int i = blockIdx.x * 256 + threadIdx.x;
  if (i < n) d[i] = (_Float16)s[i];
}

__global__ __launch_bounds__(256) void k_repackT(const float* __restrict__ src,
                                                 _Float16* __restrict__ dst,
                                                 int K, int N) {
  int i = blockIdx.x * 256 + threadIdx.x;
  if (i >= K * N) return;
  int n = i / K, k = i % K;
  dst[i] = (_Float16)src[(size_t)k * N + n];
}

__global__ __launch_bounds__(256) void k_proto_norm(const float* __restrict__ proto,
                                                    _Float16* __restrict__ Pn) {
  int wave = threadIdx.x >> 5, lane = threadIdx.x & 31;
  int j = blockIdx.x * 8 + wave;
  if (j >= PNPAD_) return;
  _Float16* outp = Pn + (size_t)j * D_;
  if (j >= P_ * NP_) {
    for (int i = lane; i < D_; i += 32) outp[i] = (_Float16)0.f;
    return;
  }
  int p = j >> 2, kk = j & 3;
  const float* base = proto + (size_t)p * D_ * NP_ + kk;
  float v[12]; float ss = 0.f;
#pragma unroll
  for (int i = 0; i < 12; ++i) {
    int d = lane + i * 32;
    v[i] = base[(size_t)d * NP_];
    ss += v[i] * v[i];
  }
  for (int m = 16; m >= 1; m >>= 1) ss += __shfl_xor(ss, m, 32);
  float inv = 1.f / fmaxf(sqrtf(ss), 1e-12f);
#pragma unroll
  for (int i = 0; i < 12; ++i) outp[lane + i * 32] = (_Float16)(v[i] * inv);
}

// ---------------- patch embed ----------------

__global__ __launch_bounds__(256) void k_im2col(const float* __restrict__ x,
                                                float* __restrict__ Xc) {
  int i = blockIdx.x * 256 + threadIdx.x;
  if (i >= B_ * N_ * 768) return;
  int col = i % 768, rowi = i / 768;
  int n = rowi % N_, b = rowi / N_;
  int c = col / 256, rem = col % 256, ii = rem / 16, jj = rem % 16;
  int py = n / 14, px = n % 14;
  Xc[i] = x[(((size_t)b * 3 + c) * 224 + py * 16 + ii) * 224 + px * 16 + jj];
}

__global__ __launch_bounds__(384) void k_assemble(const float* __restrict__ Cpe,
                                                  const float* __restrict__ cls_tok,
                                                  const float* __restrict__ pos,
                                                  float* __restrict__ tok) {
  int t = blockIdx.x, b = blockIdx.y, d = threadIdx.x;
  float v = (t == 0) ? cls_tok[d] : Cpe[((size_t)b * N_ + (t - 1)) * D_ + d];
  tok[((size_t)b * T_ + t) * D_ + d] = v + pos[(size_t)t * D_ + d];
}

// ---------------- WMMA GEMM, double-buffered, async B staging ----------------
// C[M,N] = epilogue(A[M,K]fp32 @ Bt[N,K]f16 + bias)
// mode 0: +bias ; 1: +bias+Res ; 2: +bias,GELU ; 3: proto-dist scatter
// Block tile 128(M) x 64(N); 8 waves as 4(M) x 2(N), each wave 32x32 (2x2 WMMA).
__global__ __launch_bounds__(256)
void k_gemm(const float* __restrict__ A, const _Float16* __restrict__ Bt,
            const float* __restrict__ bias, const float* __restrict__ Res,
            float* __restrict__ C, int M, int N, int K, int mode,
            long aStrideZ, long cStrideZ) {
  __shared__ _Float16 As[2][128 * 32];
  __shared__ _Float16 Bs[2][64 * 32];
  const float* Az = A + (size_t)blockIdx.z * (size_t)aStrideZ;
  float*       Cz = C + (size_t)blockIdx.z * (size_t)cStrideZ;
  const int nBase = blockIdx.x * 64;
  const int mBase = blockIdx.y * 128;
  const int tid = threadIdx.x;
  const int wave = tid >> 5, lane = tid & 31;
  const int wm = wave & 3, wn = wave >> 2;
  const bool mfull = (mBase + 128 <= M);

  // A staging: 128x32 fp32 = 1024 float4; 4 per thread.
  const int ar = tid >> 3;                 // base row (thread covers rows ar, ar+32, ...)
  const int ac4 = (tid & 7) << 2;          // col within 32
  // B staging: 64x32 halves = 256 x 16B; 1 per thread.
  const int br = tid >> 2;
  const int bc8 = (tid & 3) << 3;

  float4 areg[4];
#if !USE_ASYNC_LDS
  uint4 breg;
#endif

  auto issueA = [&](int k0) {
#pragma unroll
    for (int j = 0; j < 4; ++j) {
      int grow = mBase + ar + 32 * j;
      if (mfull || grow < M)
        areg[j] = *(const float4*)(Az + (size_t)grow * K + k0 + ac4);
      else
        areg[j] = make_float4(0.f, 0.f, 0.f, 0.f);
    }
  };
  auto storeA = [&](int buf) {
#pragma unroll
    for (int j = 0; j < 4; ++j) {
      union { _Float16 h[4]; uint2 u; } pk;
      pk.h[0] = (_Float16)areg[j].x; pk.h[1] = (_Float16)areg[j].y;
      pk.h[2] = (_Float16)areg[j].z; pk.h[3] = (_Float16)areg[j].w;
      *(uint2*)&As[buf][(ar + 32 * j) * 32 + ac4] = pk.u;
    }
  };
  auto issueB = [&](int k0, int buf) {
    const void* g = Bt + (size_t)(nBase + br) * K + k0 + bc8;
#if USE_ASYNC_LDS
    async_b128(&Bs[buf][br * 32 + bc8], g);
#else
    breg = *(const uint4*)g;
#endif
  };
  auto storeB = [&](int buf) {
#if !USE_ASYNC_LDS
    *(uint4*)&Bs[buf][br * 32 + bc8] = breg;
#endif
  };

  v8f zero = {};
  v8f acc[2][2] = {{zero, zero}, {zero, zero}};
  const int nk = K >> 5;

  issueA(0); issueB(0, 0);
  storeA(0); storeB(0);
  wait_async0();
  __syncthreads();

  for (int kk = 0; kk < nk; ++kk) {
    const int cur = kk & 1, nxt = cur ^ 1;
    if (kk + 1 < nk) { issueA((kk + 1) << 5); issueB((kk + 1) << 5, nxt); }
    v16h a0 = load_fragA(As[cur] + (wm * 32 + 0)  * 32, 32);
    v16h a1 = load_fragA(As[cur] + (wm * 32 + 16) * 32, 32);
    v16h b0 = load_fragB(Bs[cur] + (wn * 32 + 0)  * 32, 32);
    v16h b1 = load_fragB(Bs[cur] + (wn * 32 + 16) * 32, 32);
    acc[0][0] = wmma_f32_f16(a0, b0, acc[0][0]);
    acc[0][1] = wmma_f32_f16(a0, b1, acc[0][1]);
    acc[1][0] = wmma_f32_f16(a1, b0, acc[1][0]);
    acc[1][1] = wmma_f32_f16(a1, b1, acc[1][1]);
    if (kk + 1 < nk) { storeA(nxt); storeB(nxt); wait_async0(); }
    __syncthreads();
  }

  const int lane15 = lane & 15;
  const int rbase = (lane >> 4) << 3;
#pragma unroll
  for (int jm = 0; jm < 2; ++jm) {
#pragma unroll
    for (int jn = 0; jn < 2; ++jn) {
      int col = nBase + wn * 32 + jn * 16 + lane15;
#pragma unroll
      for (int r = 0; r < 8; ++r) {
        int row = mBase + wm * 32 + jm * 16 + rbase + r;
        float v = acc[jm][jn][r];
        if (mode == 3) {
          if (row < M && col < P_ * NP_)
            Cz[(size_t)(col >> 2) * (N_ * NP_) + row * NP_ + (col & 3)] = v;
        } else if (row < M) {
          v += bias[col];
          if (mode == 1) v += Res[(size_t)row * N + col];
          else if (mode == 2) v = 0.5f * v * (1.f + erff(v * 0.70710678f));
          Cz[(size_t)row * N + col] = v;
        }
      }
    }
  }
}

// ---------------- layernorm (wave per row) ----------------

__global__ __launch_bounds__(256)
void k_layernorm(const float* __restrict__ X, const float* __restrict__ S,
                 const float* __restrict__ Bv, float* __restrict__ Y, int Mrows) {
  int wave = threadIdx.x >> 5, lane = threadIdx.x & 31;
  int row = blockIdx.x * 8 + wave;
  if (row >= Mrows) return;
  const float* xr = X + (size_t)row * D_;
  float v[12]; float s = 0.f;
#pragma unroll
  for (int i = 0; i < 12; ++i) { v[i] = xr[lane + i * 32]; s += v[i]; }
  for (int m = 16; m >= 1; m >>= 1) s += __shfl_xor(s, m, 32);
  float mean = s * (1.f / D_);
  float vs = 0.f;
#pragma unroll
  for (int i = 0; i < 12; ++i) { float d = v[i] - mean; vs += d * d; }
  for (int m = 16; m >= 1; m >>= 1) vs += __shfl_xor(vs, m, 32);
  float inv = rsqrtf(vs * (1.f / D_) + 1e-6f);
  float* yr = Y + (size_t)row * D_;
#pragma unroll
  for (int i = 0; i < 12; ++i) {
    int c = lane + i * 32;
    yr[c] = (v[i] - mean) * inv * S[c] + Bv[c];
  }
}

// ---------------- attention ----------------

#define TPAD 208
#define KPAD 224

__global__ __launch_bounds__(256)
void k_attention(const float* __restrict__ qkv, float* __restrict__ O) {
  __shared__ _Float16 Ks[TPAD * 64];
  __shared__ _Float16 Vt[64 * KPAD];
  __shared__ _Float16 Qs[16 * 64];
  __shared__ float    Sf[16 * TPAD];
  __shared__ _Float16 Sh[16 * KPAD];
  const int qb = blockIdx.x, hh = blockIdx.y, b = blockIdx.z;
  const int tid = threadIdx.x;
  // valid rows: unconditional batched loads
  for (int i = tid; i < T_ * 64; i += 256) {
    int n = i >> 6, dd = i & 63;
    const float* rowp = qkv + (size_t)(b * T_ + n) * (3 * D_) + hh * HD_ + dd;
    Ks[n * 64 + dd] = (_Float16)rowp[D_];
    Vt[dd * KPAD + n] = (_Float16)rowp[2 * D_];
  }
  // zero pad rows/cols
  for (int i = tid; i < (TPAD - T_) * 64; i += 256) {
    int n = T_ + i / 64, dd = i & 63;
    Ks[n * 64 + dd] = (_Float16)0.f;
    Vt[dd * KPAD + n] = (_Float16)0.f;
  }
  for (int i = tid; i < 64 * (KPAD - TPAD); i += 256) {
    int dd = i / (KPAD - TPAD), n = TPAD + i % (KPAD - TPAD);
    Vt[dd * KPAD + n] = (_Float16)0.f;
  }
  for (int i = tid; i < 16 * KPAD; i += 256) Sh[i] = (_Float16)0.f;
  {
    int tlim = (qb == 12) ? (T_ - qb * 16) : 16;   // valid rows in this q-block
    for (int i = tid; i < tlim * 64; i += 256) {
      int r = i >> 6, dd = i & 63;
      Qs[i] = (_Float16)qkv[(size_t)(b * T_ + qb * 16 + r) * (3 * D_) + hh * HD_ + dd];
    }
    for (int i = tid + tlim * 64; i < 16 * 64; i += 256) Qs[i] = (_Float16)0.f;
  }
  __syncthreads();
  const int wave = tid >> 5, lane = tid & 31;
  const int lane15 = lane & 15, rbase = (lane >> 4) << 3;
  for (int jt = wave; jt < 13; jt += 8) {   // S = Q K^T * scale
    v8f acc = {};
#pragma unroll
    for (int ks = 0; ks < 2; ++ks) {
      v16h a = load_fragA(Qs + ks * 32, 64);
      v16h bm = load_fragB(Ks + jt * 16 * 64 + ks * 32, 64);
      acc = wmma_f32_f16(a, bm, acc);
    }
    int col = jt * 16 + lane15;
#pragma unroll
    for (int r = 0; r < 8; ++r) {
      float v = acc[r] * 0.125f;
      if (col >= T_) v = -1e30f;
      Sf[(rbase + r) * TPAD + col] = v;
    }
  }
  __syncthreads();
  {  // softmax, 16 lanes per row
    int r = tid >> 4, sub = tid & 15;
    float vb[13]; float mx = -1e30f;
#pragma unroll
    for (int i = 0; i < 13; ++i) {
      float v = Sf[r * TPAD + sub + i * 16];
      vb[i] = v; mx = fmaxf(mx, v);
    }
    for (int m = 8; m >= 1; m >>= 1) mx = fmaxf(mx, __shfl_xor(mx, m, 32));
    float sum = 0.f;
#pragma unroll
    for (int i = 0; i < 13; ++i) { float e = __expf(vb[i] - mx); vb[i] = e; sum += e; }
    for (int m = 8; m >= 1; m >>= 1) sum += __shfl_xor(sum, m, 32);
    float inv = 1.f / sum;
#pragma unroll
    for (int i = 0; i < 13; ++i)
      Sh[r * KPAD + sub + i * 16] = (_Float16)(vb[i] * inv);
  }
  __syncthreads();
  if (wave < 4) {  // O = Sh @ V
    v8f acc = {};
#pragma unroll
    for (int ks = 0; ks < 7; ++ks) {
      v16h a = load_fragA(Sh + ks * 32, KPAD);
      v16h bm = load_fragB(Vt + wave * 16 * KPAD + ks * 32, KPAD);
      acc = wmma_f32_f16(a, bm, acc);
    }
    int col = wave * 16 + lane15;
#pragma unroll
    for (int r = 0; r < 8; ++r) {
      int t = qb * 16 + rbase + r;
      if (t < T_) O[(size_t)(b * T_ + t) * D_ + hh * HD_ + col] = acc[r];
    }
  }
}

// ---------------- head ----------------

__global__ __launch_bounds__(384) void k_cls(const float* __restrict__ tokN,
                                             float* __restrict__ outp) {
  int b = blockIdx.x, d = threadIdx.x;
  outp[b * D_ + d] = tokN[((size_t)b * T_) * D_ + d];
}

__global__ __launch_bounds__(384) void k_fn(const float* __restrict__ tokN,
                                            float* __restrict__ Fn) {
  int n = blockIdx.x, b = blockIdx.y, d = threadIdx.x;
  __shared__ float red[12];
  float cls = tokN[((size_t)b * T_) * D_ + d];
  float v = tokN[((size_t)b * T_ + 1 + n) * D_ + d] - cls;
  float ss = v * v;
  for (int m = 16; m >= 1; m >>= 1) ss += __shfl_xor(ss, m, 32);
  int wave = d >> 5, lane = d & 31;
  if (lane == 0) red[wave] = ss;
  __syncthreads();
  float tot = 0.f;
#pragma unroll
  for (int i = 0; i < 12; ++i) tot += red[i];
  float inv = 1.f / fmaxf(sqrtf(tot), 1e-12f);
  Fn[((size_t)b * N_ + n) * D_ + d] = v * inv;
}

__global__ __launch_bounds__(256)
void k_topk(const float* __restrict__ dm, const float* __restrict__ ps,
            float* __restrict__ scores) {
  int wave = threadIdx.x >> 5, lane = threadIdx.x & 31;
  int gid = blockIdx.x * 8 + wave;
  if (gid >= B_ * P_) return;
  int p = gid % P_;
  const float* base = dm + (size_t)gid * (N_ * NP_);
  float vals[25];
#pragma unroll
  for (int i = 0; i < 25; ++i) {
    int idx = lane + i * 32;
    vals[i] = (idx < N_ * NP_) ? base[idx] : -3.0e38f;
  }
  float w[4]; float fac = 1e-10f;
#pragma unroll
  for (int k = 0; k < 4; ++k) {
    float s = 1.f / (1.f + __expf(-ps[p * NP_ + k] * 100.f));
    w[k] = s; fac += s;
  }
#pragma unroll
  for (int k = 0; k < 4; ++k) w[k] = w[k] * (float)NP_ / fac;
  float score = 0.f;
#pragma unroll
  for (int round = 0; round < 4; ++round) {
    float bv = -3.0e38f; int bi = 0x7fffffff;
#pragma unroll
    for (int i = 0; i < 25; ++i) {
      int idx = lane + i * 32;
      if (vals[i] > bv) { bv = vals[i]; bi = idx; }
    }
    for (int m = 16; m >= 1; m >>= 1) {
      float ov = __shfl_xor(bv, m, 32);
      int   oi = __shfl_xor(bi, m, 32);
      if (ov > bv || (ov == bv && oi < bi)) { bv = ov; bi = oi; }
    }
    score += bv * w[round];
    if ((bi & 31) == lane) {
#pragma unroll
      for (int i = 0; i < 25; ++i)
        if (i == (bi >> 5)) vals[i] -= 100000.0f;
    }
  }
  if (lane == 0) scores[gid] = score;
}

// ---------------- host ----------------

static inline int cdiv_(int a, int b) { return (a + b - 1) / b; }

extern "C" void kernel_launch(void* const* d_in, const int* in_sizes, int n_in,
                              void* d_out, int out_size, void* d_ws, size_t ws_size,
                              hipStream_t stream) {
  const float* x       = (const float*)d_in[0];
  const float* patch_w = (const float*)d_in[1];
  const float* patch_b = (const float*)d_in[2];
  const float* cls_tok = (const float*)d_in[3];
  const float* pos     = (const float*)d_in[4];
  const float* ln1_s   = (const float*)d_in[5];
  const float* ln1_b   = (const float*)d_in[6];
  const float* qkv_w   = (const float*)d_in[7];
  const float* qkv_b   = (const float*)d_in[8];
  const float* proj_w  = (const float*)d_in[9];
  const float* proj_b  = (const float*)d_in[10];
  const float* ln2_s   = (const float*)d_in[11];
  const float* ln2_b   = (const float*)d_in[12];
  const float* fc1_w   = (const float*)d_in[13];
  const float* fc1_b   = (const float*)d_in[14];
  const float* fc2_w   = (const float*)d_in[15];
  const float* fc2_b   = (const float*)d_in[16];
  const float* norm_s  = (const float*)d_in[17];
  const float* norm_b  = (const float*)d_in[18];
  const float* proto   = (const float*)d_in[19];
  const float* psel    = (const float*)d_in[20];
  float* outp = (float*)d_out;
  (void)in_sizes; (void)n_in; (void)out_size; (void)ws_size;

  size_t off = 0;
  char* wsb = (char*)d_ws;
  auto alloc = [&](size_t bytes) -> void* {
    void* p = wsb + off;
    off = (off + bytes + 255) & ~(size_t)255;
    return p;
  };
  float* tokA = (float*)alloc((size_t)MT_ * D_ * 4);
  float* tokB = (float*)alloc((size_t)MT_ * D_ * 4);
  float* hbuf = (float*)alloc((size_t)MT_ * D_ * 4);
  float* obuf = (float*)alloc((size_t)MT_ * D_ * 4);
  float* tokN = (float*)alloc((size_t)MT_ * D_ * 4);
  float* qkvb = (float*)alloc((size_t)MT_ * 3 * D_ * 4);
  float* gbuf = (float*)alloc((size_t)MT_ * 4 * D_ * 4);
  float* Xc   = (float*)alloc((size_t)B_ * N_ * 768 * 4);
  float* Cpe  = (float*)alloc((size_t)B_ * N_ * D_ * 4);
  float* Fn   = (float*)alloc((size_t)B_ * N_ * D_ * 4);
  float* dm   = (float*)alloc((size_t)B_ * P_ * N_ * NP_ * 4);
  _Float16* wq  = (_Float16*)alloc((size_t)DEPTH_ * 3 * D_ * D_ * 2);
  _Float16* wp  = (_Float16*)alloc((size_t)DEPTH_ * D_ * D_ * 2);
  _Float16* w1  = (_Float16*)alloc((size_t)DEPTH_ * 4 * D_ * D_ * 2);
  _Float16* w2  = (_Float16*)alloc((size_t)DEPTH_ * D_ * 4 * D_ * 2);
  _Float16* pwh = (_Float16*)alloc((size_t)D_ * 768 * 2);
  _Float16* Pn  = (_Float16*)alloc((size_t)PNPAD_ * D_ * 2);

  // ---- weight prep ----
  k_convert<<<cdiv_(D_ * 768, 256), 256, 0, stream>>>(patch_w, pwh, D_ * 768);
  for (int d = 0; d < DEPTH_; ++d) {
    k_repackT<<<cdiv_(D_ * 3 * D_, 256), 256, 0, stream>>>(
        qkv_w + (size_t)d * D_ * 3 * D_, wq + (size_t)d * 3 * D_ * D_, D_, 3 * D_);
    k_repackT<<<cdiv_(D_ * D_, 256), 256, 0, stream>>>(
        proj_w + (size_t)d * D_ * D_, wp + (size_t)d * D_ * D_, D_, D_);
    k_repackT<<<cdiv_(D_ * 4 * D_, 256), 256, 0, stream>>>(
        fc1_w + (size_t)d * D_ * 4 * D_, w1 + (size_t)d * 4 * D_ * D_, D_, 4 * D_);
    k_repackT<<<cdiv_(4 * D_ * D_, 256), 256, 0, stream>>>(
        fc2_w + (size_t)d * 4 * D_ * D_, w2 + (size_t)d * D_ * 4 * D_, 4 * D_, D_);
  }
  k_proto_norm<<<cdiv_(PNPAD_, 8), 256, 0, stream>>>(proto, Pn);

  // ---- patch embed ----
  k_im2col<<<cdiv_(B_ * N_ * 768, 256), 256, 0, stream>>>(x, Xc);
  k_gemm<<<dim3(D_ / 64, cdiv_(B_ * N_, 128), 1), 256, 0, stream>>>(
      Xc, pwh, patch_b, nullptr, Cpe, B_ * N_, D_, 768, 0, 0, 0);
  k_assemble<<<dim3(T_, B_), 384, 0, stream>>>(Cpe, cls_tok, pos, tokA);

  // ---- transformer blocks ----
  for (int d = 0; d < DEPTH_; ++d) {
    k_layernorm<<<cdiv_(MT_, 8), 256, 0, stream>>>(tokA, ln1_s + d * D_,
                                                   ln1_b + d * D_, hbuf, MT_);
    k_gemm<<<dim3(3 * D_ / 64, cdiv_(MT_, 128), 1), 256, 0, stream>>>(
        hbuf, wq + (size_t)d * 3 * D_ * D_, qkv_b + d * 3 * D_, nullptr,
        qkvb, MT_, 3 * D_, D_, 0, 0, 0);
    k_attention<<<dim3(13, H_, B_), 256, 0, stream>>>(qkvb, obuf);
    k_gemm<<<dim3(D_ / 64, cdiv_(MT_, 128), 1), 256, 0, stream>>>(
        obuf, wp + (size_t)d * D_ * D_, proj_b + d * D_, tokA,
        tokB, MT_, D_, D_, 1, 0, 0);
    k_layernorm<<<cdiv_(MT_, 8), 256, 0, stream>>>(tokB, ln2_s + d * D_,
                                                   ln2_b + d * D_, hbuf, MT_);
    k_gemm<<<dim3(4 * D_ / 64, cdiv_(MT_, 128), 1), 256, 0, stream>>>(
        hbuf, w1 + (size_t)d * 4 * D_ * D_, fc1_b + d * 4 * D_, nullptr,
        gbuf, MT_, 4 * D_, D_, 2, 0, 0);
    k_gemm<<<dim3(D_ / 64, cdiv_(MT_, 128), 1), 256, 0, stream>>>(
        gbuf, w2 + (size_t)d * D_ * 4 * D_, fc2_b + d * D_, tokB,
        tokA, MT_, D_, 4 * D_, 1, 0, 0);
  }

  // ---- head ----
  k_layernorm<<<cdiv_(MT_, 8), 256, 0, stream>>>(tokA, norm_s, norm_b, tokN, MT_);
  k_cls<<<B_, 384, 0, stream>>>(tokN, outp);
  k_fn<<<dim3(N_, B_), 384, 0, stream>>>(tokN, Fn);
  k_gemm<<<dim3(PNPAD_ / 64, cdiv_(N_, 128), B_), 256, 0, stream>>>(
      Fn, Pn, nullptr, nullptr, dm, N_, PNPAD_, D_, 3,
      (long)N_ * D_, (long)P_ * N_ * NP_);
  k_topk<<<cdiv_(B_ * P_, 8), 256, 0, stream>>>(dm, psel, outp + B_ * D_);
}
